// FlaxFalconAttention_12515534700805
// MI455X (gfx1250) — matrix operand order repaired and checked by reading, not compiled, not run on hardware
//
#include <hip/hip_runtime.h>

#define B_   2
#define S_   2048
#define D_   4096
#define NH_  32
#define NKV_ 8
#define HD_  128

typedef __attribute__((ext_vector_type(16))) __bf16 v16bf;
typedef __attribute__((ext_vector_type(8)))  float  v8f;

#define LDT 40  // LDS row stride in bf16 (80B: bank-friendly, 16B aligned)

#if __has_builtin(__builtin_amdgcn_global_load_async_to_lds_b128) && \
    __has_builtin(__builtin_amdgcn_s_wait_asynccnt)
#define HAVE_ASYNC_LDS 1
#else
#define HAVE_ASYNC_LDS 0
#endif

union BFrag { v16bf v; uint4 q[2]; };

static __device__ __forceinline__ v16bf load_frag16(const __bf16* p0, const __bf16* p1) {
  BFrag f;
  f.q[0] = *reinterpret_cast<const uint4*>(p0);
  f.q[1] = *reinterpret_cast<const uint4*>(p1);
  return f.v;
}

#if HAVE_ASYNC_LDS
typedef int vi4 __attribute__((vector_size(4 * sizeof(int))));
typedef __attribute__((address_space(1))) vi4* gvi4_ptr;
typedef __attribute__((address_space(3))) vi4* lvi4_ptr;
static __device__ __forceinline__ void async_copy16(const __bf16* g, __bf16* l) {
  __builtin_amdgcn_global_load_async_to_lds_b128(
      (gvi4_ptr)(void*)g, (lvi4_ptr)(void*)l, 0, 0);
}
static __device__ __forceinline__ void async_wait0() {
  __builtin_amdgcn_s_wait_asynccnt(0);
}
#else
static __device__ __forceinline__ void async_wait0() {}
#endif

// Stage one 128x32 bf16 tile of A and of B into LDS (per-thread: 2x16B each).
static __device__ __forceinline__ void stage_tile(
    const __bf16* __restrict__ arow, const __bf16* __restrict__ brow, int k0,
    __bf16 (&sA)[128][LDT], __bf16 (&sB)[128][LDT], int lr, int lk)
{
#if HAVE_ASYNC_LDS
  async_copy16(arow + k0 + lk,     &sA[lr][lk]);
  async_copy16(arow + k0 + lk + 8, &sA[lr][lk + 8]);
  async_copy16(brow + k0 + lk,     &sB[lr][lk]);
  async_copy16(brow + k0 + lk + 8, &sB[lr][lk + 8]);
#else
  const uint4* pa = reinterpret_cast<const uint4*>(arow + k0 + lk);
  *reinterpret_cast<uint4*>(&sA[lr][lk])     = pa[0];
  *reinterpret_cast<uint4*>(&sA[lr][lk + 8]) = pa[1];
  const uint4* pb = reinterpret_cast<const uint4*>(brow + k0 + lk);
  *reinterpret_cast<uint4*>(&sB[lr][lk])     = pb[0];
  *reinterpret_cast<uint4*>(&sB[lr][lk + 8]) = pb[1];
#endif
}

// Double-buffered 128x128 GEMM core, K=4096: 8 waves (2x4), wave = 64x32 = 4x2 WMMA.
static __device__ __forceinline__ void gemm_core(
    const __bf16* __restrict__ arow, const __bf16* __restrict__ brow,
    __bf16 (&sA)[2][128][LDT], __bf16 (&sB)[2][128][LDT],
    int tid, v8f (&acc)[8])
{
  const int lane = tid & 31, l16 = lane & 15, half = lane >> 4;
  const int wave = tid >> 5, wm = wave >> 2, wn = wave & 3;
  const int lr = tid >> 1, lk = (tid & 1) * 16;

  stage_tile(arow, brow, 0, sA[0], sB[0], lr, lk);
  async_wait0();
  __syncthreads();

  int buf = 0;
  for (int k0 = 0; k0 < 4096; k0 += 32, buf ^= 1) {
    if (k0 + 32 < 4096)
      stage_tile(arow, brow, k0 + 32, sA[buf ^ 1], sB[buf ^ 1], lr, lk);

    v16bf af[4], bfv[2];
#pragma unroll
    for (int i = 0; i < 4; ++i) {
      int r = wm * 64 + i * 16 + l16;
      af[i] = load_frag16(&sA[buf][r][half * 8], &sA[buf][r][16 + half * 8]);
    }
#pragma unroll
    for (int j = 0; j < 2; ++j) {
      int n = wn * 32 + j * 16 + l16;
      bfv[j] = load_frag16(&sB[buf][n][half * 8], &sB[buf][n][16 + half * 8]);
    }
#pragma unroll
    for (int i = 0; i < 4; ++i)
#pragma unroll
      for (int j = 0; j < 2; ++j)
        acc[i * 2 + j] = __builtin_amdgcn_wmma_f32_16x16x32_bf16(
            false, af[i], false, bfv[j], (short)0, acc[i * 2 + j], false, false);

    async_wait0();        // own async stage done before the barrier
    __syncthreads();      // single barrier per K-step (double buffered)
  }
}

// ---------------------------------------------------------------------------
// 1) fp32 -> bf16 convert (row-major preserved)
// ---------------------------------------------------------------------------
__global__ __launch_bounds__(256)
void cvt_bf16_kernel(const float* __restrict__ src, __bf16* __restrict__ dst, int n) {
  int i = (blockIdx.x * 256 + threadIdx.x) * 8;
  if (i >= n) return;
  float4 f0 = *reinterpret_cast<const float4*>(src + i);
  float4 f1 = *reinterpret_cast<const float4*>(src + i + 4);
  union { __bf16 h[8]; uint4 u; } p;
  p.h[0] = (__bf16)f0.x; p.h[1] = (__bf16)f0.y; p.h[2] = (__bf16)f0.z; p.h[3] = (__bf16)f0.w;
  p.h[4] = (__bf16)f1.x; p.h[5] = (__bf16)f1.y; p.h[6] = (__bf16)f1.z; p.h[7] = (__bf16)f1.w;
  *reinterpret_cast<uint4*>(dst + i) = p.u;
}

// ---------------------------------------------------------------------------
// 2) fp32 [K][N] -> bf16 [N][K] tiled transpose-convert
// ---------------------------------------------------------------------------
__global__ __launch_bounds__(256)
void tconv_bf16_kernel(const float* __restrict__ src, __bf16* __restrict__ dst, int K, int N) {
  __shared__ float tile[32][33];
  int n0 = blockIdx.x * 32, k0 = blockIdx.y * 32;
  int tx = threadIdx.x, ty = threadIdx.y;            // 32 x 8
#pragma unroll
  for (int i = 0; i < 32; i += 8)
    tile[ty + i][tx] = src[(size_t)(k0 + ty + i) * N + n0 + tx];
  __syncthreads();
#pragma unroll
  for (int i = 0; i < 32; i += 8)
    dst[(size_t)(n0 + ty + i) * K + k0 + tx] = (__bf16)tile[tx][ty + i];
}

// ---------------------------------------------------------------------------
// 3) QKV GEMM (bf16 WMMA, f32 acc) with fused RoPE epilogue.
// ---------------------------------------------------------------------------
__global__ __launch_bounds__(256)
void qkv_gemm_rope_kernel(const __bf16* __restrict__ A,
                          const __bf16* __restrict__ WqT,
                          const __bf16* __restrict__ WkT,
                          const __bf16* __restrict__ WvT,
                          const int*    __restrict__ pos_ids,
                          __bf16* __restrict__ Qo,   // [B][NH][S][HD]
                          __bf16* __restrict__ Ko,   // [B][NKV][S][HD]
                          __bf16* __restrict__ Vo)   // [B][NKV][S][HD]
{
  __shared__ __align__(16) __bf16 sA[2][128][LDT];
  __shared__ __align__(16) __bf16 sB[2][128][LDT];

  const int tid  = threadIdx.x;
  const int wave = tid >> 5, lane = tid & 31;
  const int l16  = lane & 15, half = lane >> 4;
  const int wm   = wave >> 2, wn = wave & 3;
  const int m_base = blockIdx.y * 128;
  const int n_base = blockIdx.x * 128;

  v8f acc[8];
#pragma unroll
  for (int i = 0; i < 8; ++i) acc[i] = (v8f){};

  const int lr = tid >> 1;
  const int ng = n_base + lr;
  const __bf16* brow;
  if (ng < 4096)      brow = WqT + (size_t)ng * 4096;
  else if (ng < 5120) brow = WkT + (size_t)(ng - 4096) * 4096;
  else                brow = WvT + (size_t)(ng - 5120) * 4096;
  const __bf16* arow = A + (size_t)(m_base + lr) * 4096;

  gemm_core(arow, brow, sA, sB, tid, acc);

  // Epilogue: RoPE (+ scale for Q), scatter to head-major bf16 buffers.
  const float qscale   = 0.08838834764831845f;     // 1/sqrt(128)
  const float logtheta = 13.122363377404328f;      // ln(500000)
#pragma unroll
  for (int i = 0; i < 4; ++i) {
#pragma unroll
    for (int j = 0; j < 2; ++j) {
      const int f    = n_base + wn * 32 + j * 16 + l16;
      const int row0 = m_base + wm * 64 + i * 16 + 8 * half;
      v8f a = acc[i * 2 + j];
      if (f < 4096) {                                     // ---- Q + RoPE + scale
        const int h = f >> 7, d = f & 127;
        const float invf = __expf(-(float)(d & ~1) * (1.0f / 128.0f) * logtheta);
#pragma unroll
        for (int r = 0; r < 8; ++r) {
          const int row = row0 + r;
          const int s = row & 2047, b = row >> 11;
          const float pos = (float)pos_ids[row];
          float v = a[r];
          float w = __shfl_xor(v, 1, 32);
          float c, sn; __sincosf(pos * invf, &sn, &c);
          float out = (d & 1) ? (w * sn + v * c) : (v * c - w * sn);
          Qo[(((size_t)b * NH_ + h) * S_ + s) * HD_ + d] = (__bf16)(out * qscale);
        }
      } else if (f < 5120) {                              // ---- K + RoPE
        const int h = (f - 4096) >> 7, d = f & 127;
        const float invf = __expf(-(float)(d & ~1) * (1.0f / 128.0f) * logtheta);
#pragma unroll
        for (int r = 0; r < 8; ++r) {
          const int row = row0 + r;
          const int s = row & 2047, b = row >> 11;
          const float pos = (float)pos_ids[row];
          float v = a[r];
          float w = __shfl_xor(v, 1, 32);
          float c, sn; __sincosf(pos * invf, &sn, &c);
          float out = (d & 1) ? (w * sn + v * c) : (v * c - w * sn);
          Ko[(((size_t)b * NKV_ + h) * S_ + s) * HD_ + d] = (__bf16)out;
        }
      } else {                                            // ---- V passthrough
        const int h = (f - 5120) >> 7, d = f & 127;
#pragma unroll
        for (int r = 0; r < 8; ++r) {
          const int row = row0 + r;
          const int s = row & 2047, b = row >> 11;
          Vo[(((size_t)b * NKV_ + h) * S_ + s) * HD_ + d] = (__bf16)a[r];
        }
      }
    }
  }
}

// ---------------------------------------------------------------------------
// 4) Flash attention: grid (S/128, NH, B), 8 waves, each wave owns 16 queries.
// ---------------------------------------------------------------------------
__global__ __launch_bounds__(256)
void flash_attn_kernel(const __bf16* __restrict__ Q, const __bf16* __restrict__ K,
                       const __bf16* __restrict__ V, const float* __restrict__ amask,
                       __bf16* __restrict__ O)      // [B][S][NH*HD]
{
  __shared__ __align__(16) __bf16 sK[32][136];      // [key][d]   (B-layout for scores)
  __shared__ __align__(16) __bf16 sV[128][LDT];     // [d][key]   (B-layout for P*V)
  __shared__ __align__(16) __bf16 sP[8][16][LDT];   // per-wave P tile, A-layout src
  __shared__ float sMB[32];

  const int qblk = blockIdx.x, h = blockIdx.y, b = blockIdx.z;
  const int kvh  = h >> 2;
  const int tid  = threadIdx.x, wave = tid >> 5, lane = tid & 31;
  const int l16  = lane & 15, half = lane >> 4;
  const int q0   = qblk * 128 + wave * 16;

  const __bf16* Qb = Q + ((size_t)b * NH_  + h)   * S_ * HD_;
  const __bf16* Kb = K + ((size_t)b * NKV_ + kvh) * S_ * HD_;
  const __bf16* Vb = V + ((size_t)b * NKV_ + kvh) * S_ * HD_;

  // Q fragments: rows q0..q0+15, k-chunks of 32 over HD=128
  v16bf qf[4];
  {
    const __bf16* qrow = Qb + (size_t)(q0 + l16) * HD_;
#pragma unroll
    for (int c = 0; c < 4; ++c)
      qf[c] = load_frag16(qrow + c * 32 + half * 8, qrow + c * 32 + 16 + half * 8);
  }

  v8f oacc[8];
#pragma unroll
  for (int t = 0; t < 8; ++t) oacc[t] = (v8f){};
  float m[8], l[8];
#pragma unroll
  for (int r = 0; r < 8; ++r) { m[r] = -1e30f; l[r] = 0.0f; }

  const int nkb = qblk * 4 + 4;                     // causal: keys <= qblk*128+127
  for (int kb = 0; kb < nkb; ++kb) {
    __syncthreads();
    {   // stage K block [32][128]
      int key = tid >> 3, doff = (tid & 7) * 16;
      const __bf16* src = Kb + (size_t)(kb * 32 + key) * HD_ + doff;
#if HAVE_ASYNC_LDS
      async_copy16(src,     &sK[key][doff]);
      async_copy16(src + 8, &sK[key][doff + 8]);
#else
      const uint4* ps = reinterpret_cast<const uint4*>(src);
      *reinterpret_cast<uint4*>(&sK[key][doff])     = ps[0];
      *reinterpret_cast<uint4*>(&sK[key][doff + 8]) = ps[1];
#endif
    }
#pragma unroll
    for (int pp = 0; pp < 2; ++pp) {  // stage V block transposed -> sV[d][key]
      int p = tid * 2 + pp;           // 0..511
      int key = p >> 4, d0 = (p & 15) * 8;
      union { uint4 u; __bf16 e[8]; } vv;
      vv.u = *reinterpret_cast<const uint4*>(Vb + (size_t)(kb * 32 + key) * HD_ + d0);
#pragma unroll
      for (int i = 0; i < 8; ++i) sV[d0 + i][key] = vv.e[i];
    }
    if (tid < 32)
      sMB[tid] = amask[(size_t)b * S_ + kb * 32 + tid] > 0.0f ? 0.0f : -1e30f;
    async_wait0();
    __syncthreads();

    if (kb * 32 > q0 + 15) continue;                // fully-masked block for this wave

    // scores: 2 key tiles x 4 d-chunks
    v8f sc[2]; sc[0] = (v8f){}; sc[1] = (v8f){};
#pragma unroll
    for (int jt = 0; jt < 2; ++jt) {
      const int key = jt * 16 + l16;
#pragma unroll
      for (int c = 0; c < 4; ++c) {
        v16bf kf = load_frag16(&sK[key][c * 32 + half * 8],
                               &sK[key][c * 32 + 16 + half * 8]);
        sc[jt] = __builtin_amdgcn_wmma_f32_16x16x32_bf16(
            false, qf[c], false, kf, (short)0, sc[jt], false, false);
      }
    }

    // online softmax
    float alpha[8];
#pragma unroll
    for (int r = 0; r < 8; ++r) {
      const int row = q0 + r + 8 * half;
      float s0 = sc[0][r] + sMB[l16];
      float s1 = sc[1][r] + sMB[16 + l16];
      if (kb * 32 + l16      > row) s0 = -1e30f;
      if (kb * 32 + 16 + l16 > row) s1 = -1e30f;
      float mx = fmaxf(s0, s1);
      mx = fmaxf(mx, __shfl_xor(mx, 1, 32));
      mx = fmaxf(mx, __shfl_xor(mx, 2, 32));
      mx = fmaxf(mx, __shfl_xor(mx, 4, 32));
      mx = fmaxf(mx, __shfl_xor(mx, 8, 32));
      const float mn = fmaxf(m[r], mx);
      const float al = __expf(m[r] - mn);
      const float p0 = __expf(s0 - mn);
      const float p1 = __expf(s1 - mn);
      float rs = p0 + p1;
      rs += __shfl_xor(rs, 1, 32);
      rs += __shfl_xor(rs, 2, 32);
      rs += __shfl_xor(rs, 4, 32);
      rs += __shfl_xor(rs, 8, 32);
      m[r] = mn; l[r] = l[r] * al + rs; alpha[r] = al;
      sP[wave][r + 8 * half][l16]      = (__bf16)p0;
      sP[wave][r + 8 * half][16 + l16] = (__bf16)p1;
    }
#pragma unroll
    for (int t = 0; t < 8; ++t)
#pragma unroll
      for (int r = 0; r < 8; ++r) oacc[t][r] *= alpha[r];

    // O += P(16x32) x V(32x128)
    v16bf pf = load_frag16(&sP[wave][l16][half * 8], &sP[wave][l16][16 + half * 8]);
#pragma unroll
    for (int t = 0; t < 8; ++t) {
      const int d = t * 16 + l16;
      v16bf vf = load_frag16(&sV[d][half * 8], &sV[d][16 + half * 8]);
      oacc[t] = __builtin_amdgcn_wmma_f32_16x16x32_bf16(
          false, pf, false, vf, (short)0, oacc[t], false, false);
    }
  }

  // normalize + write attn output bf16 [B][S][NH*HD]
#pragma unroll
  for (int t = 0; t < 8; ++t) {
#pragma unroll
    for (int r = 0; r < 8; ++r) {
      const int row = q0 + r + 8 * half;
      const float out = oacc[t][r] / fmaxf(l[r], 1e-30f);
      O[((size_t)b * S_ + row) * D_ + h * HD_ + t * 16 + l16] = (__bf16)out;
    }
  }
}

// ---------------------------------------------------------------------------
// 5) Output projection GEMM: attn bf16 [4096][4096] x Wo^T bf16 -> f32 out
// ---------------------------------------------------------------------------
__global__ __launch_bounds__(256)
void out_gemm_kernel(const __bf16* __restrict__ A, const __bf16* __restrict__ BT,
                     float* __restrict__ C)
{
  __shared__ __align__(16) __bf16 sA[2][128][LDT];
  __shared__ __align__(16) __bf16 sB[2][128][LDT];

  const int tid  = threadIdx.x;
  const int wave = tid >> 5, lane = tid & 31;
  const int l16  = lane & 15, half = lane >> 4;
  const int wm   = wave >> 2, wn = wave & 3;
  const int m_base = blockIdx.y * 128;
  const int n_base = blockIdx.x * 128;

  v8f acc[8];
#pragma unroll
  for (int i = 0; i < 8; ++i) acc[i] = (v8f){};

  const int lr = tid >> 1;
  const __bf16* arow = A  + (size_t)(m_base + lr) * 4096;
  const __bf16* brow = BT + (size_t)(n_base + lr) * 4096;

  gemm_core(arow, brow, sA, sB, tid, acc);

#pragma unroll
  for (int i = 0; i < 4; ++i)
#pragma unroll
    for (int j = 0; j < 2; ++j) {
      const int col  = n_base + wn * 32 + j * 16 + l16;
      const int row0 = m_base + wm * 64 + i * 16 + 8 * half;
#pragma unroll
      for (int r = 0; r < 8; ++r)
        C[(size_t)(row0 + r) * 4096 + col] = acc[i * 2 + j][r];
    }
}

// ---------------------------------------------------------------------------
// Launch
// ---------------------------------------------------------------------------
extern "C" void kernel_launch(void* const* d_in, const int* in_sizes, int n_in,
                              void* d_out, int out_size, void* d_ws, size_t ws_size,
                              hipStream_t stream) {
  const float* hs    = (const float*)d_in[0];
  const float* amask = (const float*)d_in[1];
  const int*   pos   = (const int*)  d_in[2];
  const float* wq    = (const float*)d_in[3];
  const float* wk    = (const float*)d_in[4];
  const float* wv    = (const float*)d_in[5];
  const float* wo    = (const float*)d_in[6];
  float* out = (float*)d_out;
  char*  ws  = (char*)d_ws;

  // workspace layout (bytes)
  __bf16* hsb  = (__bf16*)(ws + 0);          // 33554432  hidden bf16 [4096][4096]
  __bf16* wqT  = (__bf16*)(ws + 33554432);   // 33554432  Wq^T
  __bf16* wkT  = (__bf16*)(ws + 67108864);   //  8388608  Wk^T
  __bf16* wvT  = (__bf16*)(ws + 75497472);   //  8388608  Wv^T
  __bf16* woT  = (__bf16*)(ws + 83886080);   // 33554432  Wo^T
  __bf16* Qb   = (__bf16*)(ws + 117440512);  // 33554432  Q  [B][NH][S][HD]
  __bf16* Kb   = (__bf16*)(ws + 150994944);  //  8388608  K  [B][NKV][S][HD]
  __bf16* Vb   = (__bf16*)(ws + 159383552);  //  8388608  V  [B][NKV][S][HD]
  __bf16* attn = (__bf16*)(ws + 167772160);  // 33554432  attn bf16 [4096][4096]

  cvt_bf16_kernel<<<8192, 256, 0, stream>>>(hs, hsb, 16777216);

  dim3 tb(32, 8);
  tconv_bf16_kernel<<<dim3(128, 128), tb, 0, stream>>>(wq, wqT, 4096, 4096);
  tconv_bf16_kernel<<<dim3(32, 128),  tb, 0, stream>>>(wk, wkT, 4096, 1024);
  tconv_bf16_kernel<<<dim3(32, 128),  tb, 0, stream>>>(wv, wvT, 4096, 1024);
  tconv_bf16_kernel<<<dim3(128, 128), tb, 0, stream>>>(wo, woT, 4096, 4096);

  qkv_gemm_rope_kernel<<<dim3(6144 / 128, 4096 / 128), 256, 0, stream>>>(
      hsb, wqT, wkT, wvT, pos, Qb, Kb, Vb);

  flash_attn_kernel<<<dim3(S_ / 128, NH_, B_), 256, 0, stream>>>(
      Qb, Kb, Vb, amask, attn);

  out_gemm_kernel<<<dim3(4096 / 128, 4096 / 128), 256, 0, stream>>>(attn, woT, out);
}
